// SiglipSdpaAttention_34127810134493
// MI455X (gfx1250) — compile-verified, matrix operand order checked by reading, not run on hardware
//
#include <hip/hip_runtime.h>
#include <hip/hip_bf16.h>

// ---------------------------------------------------------------------------
// SigLIP-style attention with token merging, CDNA5 (gfx1250) WMMA bf16 path.
// B=32, N=1024, D=1152, R=4, M=256.
// Key trick: merge(x)@W^T + b == merge(x@W^T + b)  (merge is a mean, proj is
// affine), so K_m/V_m are computed from pre-merged x (4x less GEMM work).
// ---------------------------------------------------------------------------

#define BB 32
#define NN 1024
#define DD 1152
#define RR 4
#define MM 256   // NN/RR

typedef __attribute__((ext_vector_type(8)))  __bf16 v8bf;
typedef __attribute__((ext_vector_type(16))) __bf16 v16bf;
typedef __attribute__((ext_vector_type(8)))  float  v8f;

__device__ __forceinline__ unsigned short f2bf(float f) {
    unsigned int u = __builtin_bit_cast(unsigned int, f);
    u += 0x7FFFu + ((u >> 16) & 1u);           // round-to-nearest-even
    return (unsigned short)(u >> 16);
}

// Load one 16x32 bf16 fragment (A layout) or 32x16 fragment (B layout, from a
// [16, K] row-major "B-transposed" view) following the CDNA5 16-bit WMMA VGPR
// layout: lanes 0-15 row=lane holds K {k0..k0+7, k0+16..k0+23}; lanes 16-31
// row=lane-16 holds K {k0+8..k0+15, k0+24..k0+31}. Two 16B loads per lane.
__device__ __forceinline__ v16bf load_frag(const unsigned short* base,
                                           int stride, int k0, int lane) {
    int rs = lane & 15;
    int hi = (lane >> 4) & 1;
    const __bf16* p = (const __bf16*)(base + (size_t)rs * stride + k0 + hi * 8);
    v8bf lo = *(const v8bf*)p;
    v8bf hh = *(const v8bf*)(p + 16);
    return __builtin_shufflevector(lo, hh, 0, 1, 2, 3, 4, 5, 6, 7,
                                           8, 9, 10, 11, 12, 13, 14, 15);
}

__device__ __forceinline__ v8f wmma_bf16(v16bf a, v16bf b, v8f c) {
    return __builtin_amdgcn_wmma_f32_16x16x32_bf16(
        /*neg_a=*/false, a, /*neg_b=*/false, b,
        /*c_mod=*/(short)0, c, /*reuse_a=*/false, /*reuse_b=*/false);
}

// ---------------------------------------------------------------------------
// Kernel: fp32 -> bf16 convert (flat)
// ---------------------------------------------------------------------------
__global__ __launch_bounds__(256) void cvt_bf16_kernel(
    const float* __restrict__ in, unsigned short* __restrict__ out, int n) {
    int i = blockIdx.x * 256 + threadIdx.x;
    if (i < n) out[i] = f2bf(in[i]);
}

// ---------------------------------------------------------------------------
// Kernel: window-mean over RR tokens, fp32 in -> bf16 out. out is [B, M, D].
// ---------------------------------------------------------------------------
__global__ __launch_bounds__(256) void merge_x_kernel(
    const float* __restrict__ x, unsigned short* __restrict__ xm) {
    int i = blockIdx.x * 256 + threadIdx.x;      // over B*M*D
    if (i >= BB * MM * DD) return;
    int d = i % DD;
    int t = i / DD;
    int m = t % MM;
    int b = t / MM;
    const float* p = x + ((size_t)b * NN + (size_t)m * RR) * DD + d;
    float s = p[0] + p[DD] + p[2 * DD] + p[3 * DD];
    xm[i] = f2bf(s * 0.25f);
}

// ---------------------------------------------------------------------------
// Kernel: C[rows, cols] = A[rows, K] * W[cols, K]^T + bias
// Wave tile: 32 rows x 64 cols (8 accumulators; B fragments reused across two
// A-row fragments -> 8 WMMA per 6 fragment loads per k-step).
// Block = 8 waves = 256 rows x 64 cols. Grid = (rows/256, cols/64).
// Output: fp32 (outF) or bf16 (outB); bf16 optionally stored per-batch
// transposed: outB[(b*cols + col)*Mb + m] with b = row/Mb, m = row%Mb.
// ---------------------------------------------------------------------------
__global__ __launch_bounds__(256) void gemm_bf16_kernel(
    const unsigned short* __restrict__ A,
    const unsigned short* __restrict__ W,
    const float* __restrict__ bias,
    float* __restrict__ outF,
    unsigned short* __restrict__ outB,
    int K, int cols, int Mb, int transposeB) {
    int wave = threadIdx.x >> 5;
    int lane = threadIdx.x & 31;
    int rowbase = (blockIdx.x * 8 + wave) * 32;
    int colbase = blockIdx.y * 64;

    v8f acc00 = {}, acc01 = {}, acc02 = {}, acc03 = {};
    v8f acc10 = {}, acc11 = {}, acc12 = {}, acc13 = {};
    const unsigned short* aBase0 = A + (size_t)rowbase * K;
    const unsigned short* aBase1 = A + (size_t)(rowbase + 16) * K;
    const unsigned short* w0 = W + (size_t)(colbase +  0) * K;
    const unsigned short* w1 = W + (size_t)(colbase + 16) * K;
    const unsigned short* w2 = W + (size_t)(colbase + 32) * K;
    const unsigned short* w3 = W + (size_t)(colbase + 48) * K;

    for (int k0 = 0; k0 < K; k0 += 32) {
        v16bf a0 = load_frag(aBase0, K, k0, lane);
        v16bf a1 = load_frag(aBase1, K, k0, lane);
        v16bf b0 = load_frag(w0, K, k0, lane);
        v16bf b1 = load_frag(w1, K, k0, lane);
        v16bf b2 = load_frag(w2, K, k0, lane);
        v16bf b3 = load_frag(w3, K, k0, lane);
        acc00 = wmma_bf16(a0, b0, acc00);
        acc10 = wmma_bf16(a1, b0, acc10);
        acc01 = wmma_bf16(a0, b1, acc01);
        acc11 = wmma_bf16(a1, b1, acc11);
        acc02 = wmma_bf16(a0, b2, acc02);
        acc12 = wmma_bf16(a1, b2, acc12);
        acc03 = wmma_bf16(a0, b3, acc03);
        acc13 = wmma_bf16(a1, b3, acc13);
    }

    int cs = lane & 15;
    int rh = (lane >> 4) ? 8 : 0;
    v8f accs[2][4] = {{acc00, acc01, acc02, acc03},
                      {acc10, acc11, acc12, acc13}};
#pragma unroll
    for (int rr = 0; rr < 2; ++rr) {
#pragma unroll
        for (int t = 0; t < 4; ++t) {
            int col = colbase + t * 16 + cs;
            float bvv = bias ? bias[col] : 0.0f;
#pragma unroll
            for (int i = 0; i < 8; ++i) {
                int row = rowbase + rr * 16 + rh + i;
                float val = accs[rr][t][i] + bvv;
                if (outF) {
                    outF[(size_t)row * cols + col] = val;
                } else if (transposeB) {
                    int b = row / Mb;
                    int m = row - b * Mb;
                    outB[((size_t)b * cols + col) * Mb + m] = f2bf(val);
                } else {
                    outB[(size_t)row * cols + col] = f2bf(val);
                }
            }
        }
    }
}

// ---------------------------------------------------------------------------
// Kernel: fused attention for one (batch, 16-query tile).
//   scores = Q_tile @ K_m^T * scale + log(R); softmax; out = w @ V_m.
// Q bf16 [B,N,D]; Km bf16 [B,M,D]; VmT bf16 [B,D,M]; out bf16 [B,N,D].
// ---------------------------------------------------------------------------
__global__ __launch_bounds__(256) void attn_kernel(
    const unsigned short* __restrict__ Q,
    const unsigned short* __restrict__ Km,
    const unsigned short* __restrict__ VmT,
    unsigned short* __restrict__ outAttn) {
    __shared__ unsigned short qt[16 * DD];        // 36 KB
    __shared__ float sc[16 * MM];                 // 16 KB
    __shared__ unsigned short wb[16 * MM];        // 8 KB
    __shared__ float redA[16 * 16];
    __shared__ float rowStat[16];

    int tid  = threadIdx.x;
    int wave = tid >> 5;
    int lane = tid & 31;
    int b     = blockIdx.x >> 6;                  // N/16 = 64 tiles per batch
    int qtile = blockIdx.x & 63;
    int qbase = qtile * 16;

    // Stage Q tile to LDS (16B vector copies).
    const unsigned short* qsrc = Q + ((size_t)b * NN + qbase) * DD;
    for (int i = tid; i < 16 * DD / 8; i += 256)
        ((uint4*)qt)[i] = ((const uint4*)qsrc)[i];
    __syncthreads();

    // Phase 1: scores via WMMA. 16 m-tiles over 8 waves; each wave does its
    // two m-tiles in one k-loop, sharing the Q A-fragment.
    const float scale = 0.029462783f;             // 1/sqrt(1152)
    const float lbias = 1.3862944f;               // log(4)
    {
        int mt0 = wave * 2, mt1 = wave * 2 + 1;
        v8f acc0 = {}, acc1 = {};
        const unsigned short* kb0 = Km + ((size_t)b * MM + mt0 * 16) * DD;
        const unsigned short* kb1 = Km + ((size_t)b * MM + mt1 * 16) * DD;
        for (int k0 = 0; k0 < DD; k0 += 32) {
            v16bf af  = load_frag(qt, DD, k0, lane);
            v16bf bf0 = load_frag(kb0, DD, k0, lane);
            v16bf bf1 = load_frag(kb1, DD, k0, lane);
            acc0 = wmma_bf16(af, bf0, acc0);
            acc1 = wmma_bf16(af, bf1, acc1);
        }
        int cs = lane & 15;
        int rh = (lane >> 4) ? 8 : 0;
#pragma unroll
        for (int i = 0; i < 8; ++i) {
            sc[(rh + i) * MM + mt0 * 16 + cs] = acc0[i] * scale + lbias;
            sc[(rh + i) * MM + mt1 * 16 + cs] = acc1[i] * scale + lbias;
        }
    }
    __syncthreads();

    // Phase 2: row softmax over M=256. 16 threads per row, 16 cols each.
    {
        int r = tid >> 4, cg = tid & 15;
        float mx = -1e30f;
        for (int c = cg * 16; c < cg * 16 + 16; ++c)
            mx = fmaxf(mx, sc[r * MM + c]);
        redA[r * 16 + cg] = mx;
        __syncthreads();
        if (cg == 0) {
            float m2 = redA[r * 16];
            for (int j = 1; j < 16; ++j) m2 = fmaxf(m2, redA[r * 16 + j]);
            rowStat[r] = m2;
        }
        __syncthreads();
        float rm = rowStat[r];
        float s = 0.0f;
        for (int c = cg * 16; c < cg * 16 + 16; ++c) {
            float e = __expf(sc[r * MM + c] - rm);
            sc[r * MM + c] = e;
            s += e;
        }
        redA[r * 16 + cg] = s;
        __syncthreads();
        if (cg == 0) {
            float t = 0.0f;
            for (int j = 0; j < 16; ++j) t += redA[r * 16 + j];
            rowStat[r] = t;
        }
        __syncthreads();
        float inv = 1.0f / rowStat[r];
        for (int c = cg * 16; c < cg * 16 + 16; ++c)
            wb[r * MM + c] = f2bf(sc[r * MM + c] * inv);
    }
    __syncthreads();

    // Phase 3: out = w @ V_m. 72 d-tiles over 8 waves (9 each): 4 pairs
    // sharing the w A-fragment + 1 leftover tile. K-loop M/32 = 8.
    int cs = lane & 15;
    int rh = (lane >> 4) ? 8 : 0;
#pragma unroll
    for (int j = 0; j < 4; ++j) {
        int dt0 = wave + 16 * j;
        int dt1 = dt0 + 8;
        v8f acc0 = {}, acc1 = {};
        const unsigned short* vb0 = VmT + ((size_t)b * DD + dt0 * 16) * MM;
        const unsigned short* vb1 = VmT + ((size_t)b * DD + dt1 * 16) * MM;
        for (int k0 = 0; k0 < MM; k0 += 32) {
            v16bf af  = load_frag(wb, MM, k0, lane);
            v16bf bf0 = load_frag(vb0, MM, k0, lane);
            v16bf bf1 = load_frag(vb1, MM, k0, lane);
            acc0 = wmma_bf16(af, bf0, acc0);
            acc1 = wmma_bf16(af, bf1, acc1);
        }
#pragma unroll
        for (int i = 0; i < 8; ++i) {
            size_t rowoff = ((size_t)b * NN + qbase + rh + i) * DD;
            outAttn[rowoff + dt0 * 16 + cs] = f2bf(acc0[i]);
            outAttn[rowoff + dt1 * 16 + cs] = f2bf(acc1[i]);
        }
    }
    {   // leftover d-tile: dt = 64 + wave
        int dt = 64 + wave;
        v8f acc = {};
        const unsigned short* vb = VmT + ((size_t)b * DD + dt * 16) * MM;
        for (int k0 = 0; k0 < MM; k0 += 32) {
            v16bf af  = load_frag(wb, MM, k0, lane);
            v16bf bfb = load_frag(vb, MM, k0, lane);
            acc = wmma_bf16(af, bfb, acc);
        }
#pragma unroll
        for (int i = 0; i < 8; ++i)
            outAttn[((size_t)b * NN + qbase + rh + i) * DD + dt * 16 + cs] =
                f2bf(acc[i]);
    }
}

// ---------------------------------------------------------------------------
// Host-side launch
// ---------------------------------------------------------------------------
extern "C" void kernel_launch(void* const* d_in, const int* in_sizes, int n_in,
                              void* d_out, int out_size, void* d_ws, size_t ws_size,
                              hipStream_t stream) {
    (void)in_sizes; (void)n_in; (void)out_size; (void)ws_size;

    const float* x  = (const float*)d_in[0];
    const float* Wq = (const float*)d_in[1];
    const float* bq = (const float*)d_in[2];
    const float* Wk = (const float*)d_in[3];
    const float* bk = (const float*)d_in[4];
    const float* Wv = (const float*)d_in[5];
    const float* bv = (const float*)d_in[6];
    const float* Wo = (const float*)d_in[7];
    const float* bo = (const float*)d_in[8];
    float* out = (float*)d_out;

    // Workspace layout (bytes). attn output aliases xb (xb dead by then).
    char* ws = (char*)d_ws;
    const size_t szXB  = (size_t)BB * NN * DD * 2;  // 75.5 MB bf16 x / attn
    const size_t szXM  = (size_t)BB * MM * DD * 2;  // 18.9 MB
    const size_t szQ   = (size_t)BB * NN * DD * 2;  // 75.5 MB
    const size_t szKM  = (size_t)BB * MM * DD * 2;
    const size_t szVMT = (size_t)BB * DD * MM * 2;

    unsigned short* xb   = (unsigned short*)(ws);
    unsigned short* xmb  = (unsigned short*)(ws + szXB);
    unsigned short* qb   = (unsigned short*)(ws + szXB + szXM);
    unsigned short* kmb  = (unsigned short*)(ws + szXB + szXM + szQ);
    unsigned short* vmtb = (unsigned short*)(ws + szXB + szXM + szQ + szKM);
    unsigned short* wqb  = (unsigned short*)(ws + szXB + szXM + szQ + szKM + szVMT);
    unsigned short* wkb  = wqb + (size_t)DD * DD;
    unsigned short* wvb  = wkb + (size_t)DD * DD;
    unsigned short* wob  = wvb + (size_t)DD * DD;
    unsigned short* attnb = xb;                    // alias

    const int nX  = BB * NN * DD;
    const int nW  = DD * DD;
    const int nXM = BB * MM * DD;

    // 1) Converts + merge.
    cvt_bf16_kernel<<<(nX + 255) / 256, 256, 0, stream>>>(x, xb, nX);
    merge_x_kernel<<<(nXM + 255) / 256, 256, 0, stream>>>(x, xmb);
    cvt_bf16_kernel<<<(nW + 255) / 256, 256, 0, stream>>>(Wq, wqb, nW);
    cvt_bf16_kernel<<<(nW + 255) / 256, 256, 0, stream>>>(Wk, wkb, nW);
    cvt_bf16_kernel<<<(nW + 255) / 256, 256, 0, stream>>>(Wv, wvb, nW);
    cvt_bf16_kernel<<<(nW + 255) / 256, 256, 0, stream>>>(Wo, wob, nW);

    // 2) Projections (WMMA GEMMs). Wave tile 32x64, block tile 256x64.
    dim3 gQ(BB * NN / 256, DD / 64);   // 128 x 18
    dim3 gM(BB * MM / 256, DD / 64);   // 32 x 18
    gemm_bf16_kernel<<<gQ, 256, 0, stream>>>(xb,  wqb, bq, nullptr, qb,   DD, DD, MM, 0);
    gemm_bf16_kernel<<<gM, 256, 0, stream>>>(xmb, wkb, bk, nullptr, kmb,  DD, DD, MM, 0);
    gemm_bf16_kernel<<<gM, 256, 0, stream>>>(xmb, wvb, bv, nullptr, vmtb, DD, DD, MM, 1);

    // 3) Fused attention (scores + softmax + weighted sum), WMMA both GEMMs.
    attn_kernel<<<BB * (NN / 16), 256, 0, stream>>>(qb, kmb, vmtb, attnb);

    // 4) Output projection, fp32 result.
    gemm_bf16_kernel<<<gQ, 256, 0, stream>>>(attnb, wob, bo, out, nullptr, DD, DD, MM, 0);
}
